// GraphDiTUnit_54039278519007
// MI455X (gfx1250) — matrix operand order repaired and checked by reading, not compiled, not run on hardware
//
#include <hip/hip_runtime.h>
#include <hip/hip_bf16.h>
#include <math.h>

typedef __attribute__((ext_vector_type(16))) __bf16 v16bf;
typedef __attribute__((ext_vector_type(8)))  float  v8f;
typedef __attribute__((ext_vector_type(2)))  __bf16 bf16x2;

#define WMMA_BF16(a, b, c) \
  __builtin_amdgcn_wmma_f32_16x16x32_bf16(false, (a), false, (b), (short)0, (c), false, false)

static constexpr int B_ = 2048, N_ = 2, T_ = 32, D_ = 512, H_ = 8, E_ = 128;
static constexpr int S_ = N_ * T_;   // 64 tokens per batch
static constexpr int HD = D_ / H_;   // 64 per head

// ---------------------------------------------------------------------------
// Kernel 0: convert weights to bf16 and transpose: WT[m][n][k] = W[m][k][n]
// m = 0:Wq 1:Wk 2:Wv 3:Wo. B-fragments then read contiguous k-pairs per col.
// ---------------------------------------------------------------------------
__global__ __launch_bounds__(256) void prep_weights(const float* __restrict__ Wq,
                                                    const float* __restrict__ Wk,
                                                    const float* __restrict__ Wv,
                                                    const float* __restrict__ Wo,
                                                    __bf16* __restrict__ WT) {
  int id = blockIdx.x * 256 + threadIdx.x;       // 4*512*512 total
  int m = id >> 18;
  int rem = id & 262143;
  int k = rem >> 9;
  int n = rem & 511;
  const float* W = (m == 0) ? Wq : (m == 1) ? Wk : (m == 2) ? Wv : Wo;
  WT[(size_t)m * 262144 + (size_t)n * 512 + k] = (__bf16)W[k * 512 + n];
}

// ---------------------------------------------------------------------------
// Kernel 1: edge-conditioned vectors per batch.
// ---------------------------------------------------------------------------
__global__ __launch_bounds__(256) void edge_kernel(
    const float* __restrict__ edge,
    const float* __restrict__ Wgate, const float* __restrict__ bgate,
    const float* __restrict__ Wscale, const float* __restrict__ bscale,
    const float* __restrict__ Wshift, const float* __restrict__ bshift,
    const float* __restrict__ Whs, const float* __restrict__ bhs,
    const float* __restrict__ Wbias, const float* __restrict__ bbias,
    float* __restrict__ gate, float* __restrict__ scale1,
    float* __restrict__ shift, float* __restrict__ hs1,
    float* __restrict__ ebias) {
  int b = blockIdx.x, tid = threadIdx.x;
  __shared__ float e[E_];
  if (tid < E_) e[tid] = edge[b * E_ + tid];
  __syncthreads();
  for (int d = tid; d < D_; d += 256) {
    float g = 0.f, sc = 0.f, sh = 0.f;
#pragma unroll 4
    for (int k = 0; k < E_; ++k) {
      float ev = e[k];
      g  += ev * Wgate[k * D_ + d];
      sc += ev * Wscale[k * D_ + d];
      sh += ev * Wshift[k * D_ + d];
    }
    g += bgate[d]; sc += bscale[d]; sh += bshift[d];
    gate[b * D_ + d]   = 1.0f / (1.0f + __expf(-g));
    scale1[b * D_ + d] = 1.0f + tanhf(sc);
    shift[b * D_ + d]  = sh;
  }
  if (tid < H_) {
    float a = 0.f, bb = 0.f;
    for (int k = 0; k < E_; ++k) {
      a  += e[k] * Whs[k * H_ + tid];
      bb += e[k] * Wbias[k * H_ + tid];
    }
    hs1[b * H_ + tid]   = 1.0f + tanhf(a + bhs[tid]);
    ebias[b * H_ + tid] = bb + bbias[tid];
  }
}

// 16x32 bf16 A-fragment from a row-major row pointer (16*hsel pre-folded).
__device__ inline v16bf load_a_frag(const __bf16* row_k) {
  v16bf a;
#pragma unroll
  for (int v = 0; v < 8; ++v) {
    int k0 = 2 * v;
    bf16x2 p = *(const bf16x2*)(row_k + k0);
    a[2 * v] = p.x; a[2 * v + 1] = p.y;
  }
  return a;
}

// 32x16 bf16 B-fragment: lane holds col n = lane%16; K-half pre-folded.
__device__ inline v16bf load_b_frag(const __bf16* coln_k) {
  v16bf bf;
#pragma unroll
  for (int v = 0; v < 8; ++v) {
    bf16x2 p = *(const bf16x2*)(coln_k + 2 * v);
    bf[2 * v] = p.x; bf[2 * v + 1] = p.y;
  }
  return bf;
}

// ---------------------------------------------------------------------------
// Kernel 2: FUSED QKV projection + attention + temporal mean, per batch.
// 8 waves; wave w owns head w end-to-end. 256 KB LDS:
//   xsP   : x as bf16 [64][512] during QKV, reused as P scratch (8KB/wave)
//   Qs/Ks : per-head [64 rows][64 k] bf16 (Q pre-scaled by 0.125*(1+hs))
//   Vts   : per-head transposed [64 d][64 s] bf16 (edge-modulated)
// Only HBM traffic: x read once, pooled [B,2,512] written once.
// ---------------------------------------------------------------------------
__global__ __launch_bounds__(256) void fused_kernel(
    const float* __restrict__ x, const __bf16* __restrict__ WT,
    const float* __restrict__ bq, const float* __restrict__ bk,
    const float* __restrict__ bv,
    const float* __restrict__ gate, const float* __restrict__ scale1,
    const float* __restrict__ shift, const float* __restrict__ hs1,
    const float* __restrict__ ebias, float* __restrict__ pooled) {
  int b = blockIdx.x;
  int tid = threadIdx.x;
  int w = tid >> 5;            // wave id == head id, 0..7
  int l = tid & 31;
  int lane16 = l & 15;
  int hsel = l >> 4;

  __shared__ __bf16 xsP[S_ * D_];        // 64 KB: x, later P
  __shared__ __bf16 Qs[H_][S_ * HD];     // 64 KB
  __shared__ __bf16 Ks[H_][S_ * HD];     // 64 KB
  __shared__ __bf16 Vts[H_][HD * S_];    // 64 KB

  const float* xb = x + (size_t)b * (S_ * D_);
  for (int i = tid; i < S_ * D_; i += 256) xsP[i] = (__bf16)xb[i];
  __syncthreads();

  v8f zero = {0.f, 0.f, 0.f, 0.f, 0.f, 0.f, 0.f, 0.f};

  // ---- Phase 1: QKV projection for head-w's 64-column slice -------------
  for (int mat = 0; mat < 3; ++mat) {
    const __bf16* Wt = WT + (size_t)mat * (D_ * D_);
    v8f acc[4][4];
#pragma unroll
    for (int i = 0; i < 4; ++i)
#pragma unroll
      for (int j = 0; j < 4; ++j) acc[i][j] = zero;

    for (int kb = 0; kb < D_; kb += 32) {
      v16bf A[4], Bf[4];
#pragma unroll
      for (int i = 0; i < 4; ++i)
        A[i] = load_a_frag(xsP + (16 * i + lane16) * D_ + kb + 16 * hsel);
#pragma unroll
      for (int j = 0; j < 4; ++j) {
        const __bf16* brow =
            Wt + (size_t)(64 * w + 16 * j + lane16) * D_ + kb + 16 * hsel;
        __builtin_prefetch(brow + 64, 0, 3);   // unconditional, speculative
        Bf[j] = load_b_frag(brow);
      }
#pragma unroll
      for (int i = 0; i < 4; ++i)
#pragma unroll
        for (int j = 0; j < 4; ++j)
          acc[i][j] = WMMA_BF16(A[i], Bf[j], acc[i][j]);
    }

    if (mat == 0) {
      float scl = 0.125f * hs1[b * H_ + w];   // 1/sqrt(hd) * (1+head_scale)
#pragma unroll
      for (int j = 0; j < 4; ++j) {
        int cl = 16 * j + lane16;
        float bias = bq[64 * w + cl];
#pragma unroll
        for (int i = 0; i < 4; ++i)
#pragma unroll
          for (int r = 0; r < 8; ++r) {
            int row = 16 * i + 8 * hsel + r;
            Qs[w][row * HD + cl] = (__bf16)((acc[i][j][r] + bias) * scl);
          }
      }
    } else if (mat == 1) {
#pragma unroll
      for (int j = 0; j < 4; ++j) {
        int cl = 16 * j + lane16;
        float bias = bk[64 * w + cl];
#pragma unroll
        for (int i = 0; i < 4; ++i)
#pragma unroll
          for (int r = 0; r < 8; ++r) {
            int row = 16 * i + 8 * hsel + r;
            Ks[w][row * HD + cl] = (__bf16)(acc[i][j][r] + bias);
          }
      }
    } else {
#pragma unroll
      for (int j = 0; j < 4; ++j) {
        int cl = 16 * j + lane16;
        int col = 64 * w + cl;
        float bias = bv[col];
        float gg = gate[b * D_ + col];
        float s1 = scale1[b * D_ + col];
        float sh = shift[b * D_ + col];
#pragma unroll
        for (int i = 0; i < 4; ++i)
#pragma unroll
          for (int r = 0; r < 8; ++r) {
            int row = 16 * i + 8 * hsel + r;
            float vv = (acc[i][j][r] + bias);
            vv = (vv * s1 + sh) * gg;
            Vts[w][cl * S_ + row] = (__bf16)vv;   // transposed [d][s]
          }
      }
    }
  }
  __syncthreads();   // all xsP (x) reads done before P overlays it

  // ---- Phase 2: attention for head w ------------------------------------
  float eb = ebias[b * H_ + w];
  __bf16* Pl = xsP + w * (S_ * S_);   // wave-private 8 KB P scratch

  v8f Sc[4][4];
#pragma unroll
  for (int i = 0; i < 4; ++i)
#pragma unroll
    for (int j = 0; j < 4; ++j) Sc[i][j] = zero;

  for (int kb = 0; kb < HD; kb += 32) {
    v16bf A[4], Bf[4];
#pragma unroll
    for (int i = 0; i < 4; ++i)
      A[i] = load_a_frag(Qs[w] + (16 * i + lane16) * HD + kb + 16 * hsel);
#pragma unroll
    for (int j = 0; j < 4; ++j)
      Bf[j] = load_b_frag(Ks[w] + (16 * j + lane16) * HD + kb + 16 * hsel);
#pragma unroll
    for (int i = 0; i < 4; ++i)
#pragma unroll
      for (int j = 0; j < 4; ++j) Sc[i][j] = WMMA_BF16(A[i], Bf[j], Sc[i][j]);
  }

  // pair-bias at k==q^32 + row softmax (xor 1/2/4/8 reduce the 16-lane group)
#pragma unroll
  for (int i = 0; i < 4; ++i) {
#pragma unroll
    for (int r = 0; r < 8; ++r) {
      int q = 16 * i + 8 * hsel + r;
      int kx = q ^ 32;
      float mx = -1e30f;
#pragma unroll
      for (int j = 0; j < 4; ++j) {
        int kcol = 16 * j + lane16;
        float s = Sc[i][j][r];
        if (kcol == kx) s += eb;
        Sc[i][j][r] = s;
        mx = fmaxf(mx, s);
      }
      mx = fmaxf(mx, __shfl_xor(mx, 1, 32));
      mx = fmaxf(mx, __shfl_xor(mx, 2, 32));
      mx = fmaxf(mx, __shfl_xor(mx, 4, 32));
      mx = fmaxf(mx, __shfl_xor(mx, 8, 32));
      float sum = 0.f;
#pragma unroll
      for (int j = 0; j < 4; ++j) {
        float p = __expf(Sc[i][j][r] - mx);
        Sc[i][j][r] = p;
        sum += p;
      }
      sum += __shfl_xor(sum, 1, 32);
      sum += __shfl_xor(sum, 2, 32);
      sum += __shfl_xor(sum, 4, 32);
      sum += __shfl_xor(sum, 8, 32);
      float inv = 1.0f / sum;
#pragma unroll
      for (int j = 0; j < 4; ++j)
        Pl[q * S_ + 16 * j + lane16] = (__bf16)(Sc[i][j][r] * inv);
    }
  }

  // O = P @ Vh  (wave-private LDS; DS ops are in-order within a wave)
  v8f O[4][4];
#pragma unroll
  for (int i = 0; i < 4; ++i)
#pragma unroll
    for (int j = 0; j < 4; ++j) O[i][j] = zero;

  for (int kb = 0; kb < S_; kb += 32) {
    v16bf A[4], Bf[4];
#pragma unroll
    for (int i = 0; i < 4; ++i)
      A[i] = load_a_frag(Pl + (16 * i + lane16) * S_ + kb + 16 * hsel);
#pragma unroll
    for (int j = 0; j < 4; ++j)
      Bf[j] = load_b_frag(Vts[w] + (16 * j + lane16) * S_ + kb + 16 * hsel);
#pragma unroll
    for (int i = 0; i < 4; ++i)
#pragma unroll
      for (int j = 0; j < 4; ++j) O[i][j] = WMMA_BF16(A[i], Bf[j], O[i][j]);
  }

  // temporal mean: rows 0..31 -> node 0, rows 32..63 -> node 1 (xor-16 merges
  // the M / M+8 lane halves of the C layout).
#pragma unroll
  for (int j = 0; j < 4; ++j) {
    float s0 = 0.f, s1 = 0.f;
#pragma unroll
    for (int r = 0; r < 8; ++r) {
      s0 += O[0][j][r] + O[1][j][r];
      s1 += O[2][j][r] + O[3][j][r];
    }
    s0 += __shfl_xor(s0, 16, 32);
    s1 += __shfl_xor(s1, 16, 32);
    if (l < 16) {
      int col = w * HD + 16 * j + lane16;
      pooled[(size_t)b * (N_ * D_) + col]      = s0 * (1.0f / 32.0f);
      pooled[(size_t)b * (N_ * D_) + D_ + col] = s1 * (1.0f / 32.0f);
    }
  }
}

// ---------------------------------------------------------------------------
// Kernel 3: output projection pooled[4096,512] @ Wo + bo -> d_out (f32).
// ---------------------------------------------------------------------------
__global__ __launch_bounds__(32) void proj_kernel(const float* __restrict__ pooled,
                                                  const __bf16* __restrict__ WoT,
                                                  const float* __restrict__ bo,
                                                  float* __restrict__ out) {
  int rb = blockIdx.x * 64;   // 4096 rows / 64
  int cb = blockIdx.y * 64;   // 512 cols / 64
  int l = threadIdx.x;
  int lane16 = l & 15, hsel = l >> 4;

  v8f zero = {0.f, 0.f, 0.f, 0.f, 0.f, 0.f, 0.f, 0.f};
  v8f acc[4][4];
#pragma unroll
  for (int i = 0; i < 4; ++i)
#pragma unroll
    for (int j = 0; j < 4; ++j) acc[i][j] = zero;

  for (int kb = 0; kb < D_; kb += 32) {
    v16bf A[4], Bf[4];
#pragma unroll
    for (int i = 0; i < 4; ++i) {
      const float* ar = pooled + (size_t)(rb + 16 * i + lane16) * D_ + kb + 16 * hsel;
      v16bf a;
#pragma unroll
      for (int v = 0; v < 8; ++v) {
        int k0 = 2 * v;
        a[2 * v]     = (__bf16)ar[k0];
        a[2 * v + 1] = (__bf16)ar[k0 + 1];
      }
      A[i] = a;
    }
#pragma unroll
    for (int j = 0; j < 4; ++j)
      Bf[j] = load_b_frag(WoT + (size_t)(cb + 16 * j + lane16) * D_ + kb + 16 * hsel);
#pragma unroll
    for (int i = 0; i < 4; ++i)
#pragma unroll
      for (int j = 0; j < 4; ++j) acc[i][j] = WMMA_BF16(A[i], Bf[j], acc[i][j]);
  }

#pragma unroll
  for (int j = 0; j < 4; ++j) {
    int col = cb + 16 * j + lane16;
    float bias = bo[col];
#pragma unroll
    for (int i = 0; i < 4; ++i)
#pragma unroll
      for (int r = 0; r < 8; ++r) {
        int row = rb + 16 * i + 8 * hsel + r;
        out[(size_t)row * D_ + col] = acc[i][j][r] + bias;
      }
  }
}

// ---------------------------------------------------------------------------
extern "C" void kernel_launch(void* const* d_in, const int* in_sizes, int n_in,
                              void* d_out, int out_size, void* d_ws, size_t ws_size,
                              hipStream_t stream) {
  const float* node   = (const float*)d_in[0];
  const float* edge   = (const float*)d_in[1];
  const float* Wq     = (const float*)d_in[2];
  const float* bq     = (const float*)d_in[3];
  const float* Wk     = (const float*)d_in[4];
  const float* bk     = (const float*)d_in[5];
  const float* Wv     = (const float*)d_in[6];
  const float* bv     = (const float*)d_in[7];
  const float* Wo     = (const float*)d_in[8];
  const float* bo     = (const float*)d_in[9];
  const float* Wbias  = (const float*)d_in[10];
  const float* bbias  = (const float*)d_in[11];
  const float* Wgate  = (const float*)d_in[12];
  const float* bgate  = (const float*)d_in[13];
  const float* Wscale = (const float*)d_in[14];
  const float* bscale = (const float*)d_in[15];
  const float* Wshift = (const float*)d_in[16];
  const float* bshift = (const float*)d_in[17];
  const float* Whs    = (const float*)d_in[18];
  const float* bhs    = (const float*)d_in[19];

  char* w = (char*)d_ws;
  __bf16* WT    = (__bf16*)w; w += (size_t)4 * 512 * 512 * 2;  // WqT,WkT,WvT,WoT
  float* gate   = (float*)w;  w += (size_t)B_ * D_ * 4;
  float* scl1   = (float*)w;  w += (size_t)B_ * D_ * 4;
  float* shft   = (float*)w;  w += (size_t)B_ * D_ * 4;
  float* hs1    = (float*)w;  w += (size_t)B_ * H_ * 4;
  float* ebias  = (float*)w;  w += (size_t)B_ * H_ * 4;
  float* pooled = (float*)w;  w += (size_t)B_ * N_ * D_ * 4;

  prep_weights<<<dim3(4096), dim3(256), 0, stream>>>(Wq, Wk, Wv, Wo, WT);
  edge_kernel<<<dim3(B_), dim3(256), 0, stream>>>(
      edge, Wgate, bgate, Wscale, bscale, Wshift, bshift, Whs, bhs, Wbias, bbias,
      gate, scl1, shft, hs1, ebias);
  fused_kernel<<<dim3(B_), dim3(256), 0, stream>>>(
      node, WT, bq, bk, bv, gate, scl1, shft, hs1, ebias, pooled);
  proj_kernel<<<dim3(64, 8), dim3(32), 0, stream>>>(
      pooled, WT + (size_t)3 * 512 * 512, bo, (float*)d_out);
}
